// SlotMemoryPhasor_48232482734961
// MI455X (gfx1250) — compile-verified, matrix operand order checked by reading, not run on hardware
//
#include <hip/hip_runtime.h>
#include <hip/hip_bf16.h>

// ---------------------------------------------------------------------------
// SlotMemoryPhasor fused kernel for MI455X (gfx1250, wave32, WMMA bf16)
//
// Per chunk (64 tokens x 512 dim), one workgroup of 256 threads (8 waves):
//   ww = softmax(X @ Wk2 + bk2)          (Wk2 = Wk @ SK^T * scale, prefolded)
//   rw = softmax(X @ Wq2 + bq2)
//   V  = X @ Wv + bv                     (stored transposed in LDS)
//   A  = tril(rw @ ww^T)
//   R  = A @ V ;  LN(R)*g+b ;  out = x + LN @ Wo + bo
// All GEMMs use v_wmma_f32_16x16x32_bf16. Wo is streamed with the
// Tensor Data Mover (tensor_load_to_lds) into LDS per K-step.
// ---------------------------------------------------------------------------

typedef __bf16 bf16_t;
typedef __attribute__((ext_vector_type(16))) __bf16 v16bf;
typedef __attribute__((ext_vector_type(8)))  __bf16 v8bf;
typedef __attribute__((ext_vector_type(4)))  __bf16 v4bf;
typedef __attribute__((ext_vector_type(8)))  float  v8f;
typedef __attribute__((ext_vector_type(4)))  float  v4f;
typedef __attribute__((ext_vector_type(4)))  unsigned int v4u;
typedef __attribute__((ext_vector_type(8)))  int    v8i;
typedef __attribute__((ext_vector_type(4)))  int    v4i;

#define DIMD     512
#define TOK      64        // tokens per chunk
#define NSLOT    64
#define NCHUNKS  1024      // B * S / TOK = 8*8192/64

// ---- LDS layout (bytes), total 151552 <= 160K addressable ------------------
#define XBF_STRIDE   264                       // halves per row (half-K: 256+8)
#define XBF_OFF      0                         // 64 x 264 bf16 = 33792
#define WWBF_OFF     33792                     // 64 x 72 bf16  = 9216
#define RWBF_OFF     43008                     // 64 x 72 bf16  = 9216
#define RBF_OFF      0                         // 64 x 520 bf16 = 66560 (overlays x/ww/rw, live later)
#define RBF_STRIDE   520
#define VT_OFF       66560                     // 512 x 72 bf16 = 73728
#define VT_STRIDE    72
#define LGWW_OFF     (VT_OFF)                  // 64x64 f32 (overlays Vt, earlier phase)
#define LGRW_OFF     (VT_OFF + 16384)
#define TDM_OFF      (VT_OFF)                  // 32768-byte Wo staging (overlays Vt, later phase)
#define ABF_OFF      140288                    // 64 x 72 bf16 = 9216
#define SCR_OFF      149504                    // 64x4x2 f32 = 2048
#define SMEM_BYTES   151552

// ---- workspace layout (bytes); requires ws_size >= ~1.45 MB ----------------
#define WS_PWV    0            // 16*32 frags * 1KB = 524288
#define WS_PWO    524288       // 524288
#define WS_PWK2   1048576      // 16*4 frags * 1KB = 65536
#define WS_PWQ2   1114112      // 65536
#define WS_BK2    1179648      // 256
#define WS_BQ2    1179904      // 256
#define WS_WK2F   1180160      // 512*64*4 = 131072
#define WS_WQ2F   1311232      // 131072

static __device__ inline v8f wmma_bf16(v16bf a, v16bf b, v8f c) {
    // D = A(16x32) * B(32x16) + C(16x16 f32)
    return __builtin_amdgcn_wmma_f32_16x16x32_bf16(
        /*neg_a=*/false, a, /*neg_b=*/false, b,
        /*c_mod=*/(short)0, c, /*reuse_a=*/false, /*reuse_b=*/false);
}

// Load a 16x32 bf16 A-style fragment (lane -> row, halves -> K with the
// j<8 : j+8 split and +8 row-half offset) from a row-major LDS matrix.
static __device__ inline v16bf load_afrag(const bf16_t* base, int strideH,
                                          int row0, int k0, int lane) {
    int r  = row0 + (lane & 15);
    int d8 = (lane < 16) ? 0 : 8;
    const v8bf* p0 = (const v8bf*)(base + r * strideH + k0 + d8);
    const v8bf* p1 = (const v8bf*)(base + r * strideH + k0 + 16 + d8);
    v8bf lo = *p0;
    v8bf hi = *p1;
    return __builtin_shufflevector(lo, hi, 0,1,2,3,4,5,6,7,8,9,10,11,12,13,14,15);
}

// B fragment from the pre-packed global weight array: 1KB per fragment,
// laid out so each lane reads 32 contiguous bytes (fully coalesced).
static __device__ inline v16bf load_bfrag_packed(const bf16_t* packed, int fragIdx, int lane) {
    const v16bf* p = (const v16bf*)(packed + (size_t)fragIdx * 512);
    return p[lane];
}

// Issue a TDM 1-row copy of nElem 4-byte words from global -> LDS.
// (clang-23 lane: 6-arg builtin with an extra int32x8 descriptor group.)
static __device__ inline void tdm_load(unsigned int lds_off, const void* gaddr,
                                       unsigned int nElem) {
    unsigned long long ga = (unsigned long long)(uintptr_t)gaddr;
    // D# group0: count=1, lds_addr, global_addr[56:0], type=2
    v4u g0 = { 1u,
               lds_off,
               (unsigned int)ga,
               (unsigned int)((ga >> 32) & 0x1FFFFFFu) | (2u << 30) };
    // D# group1: data_size=4B (code 2), tensor_dim0=nElem, tensor_dim1=1,
    // tile_dim0=nElem, tile_dim1=1, tensor_dim0_stride=nElem
    v8i g1 = { (int)(2u << 16),
               (int)((nElem & 0xFFFFu) << 16),
               (int)(((nElem >> 16) & 0xFFFFu) | (1u << 16)),
               (int)((nElem & 0xFFFFu) << 16),
               1,
               (int)nElem,
               0, 0 };
    v4i z4 = { 0, 0, 0, 0 };
    v8i z8 = { 0, 0, 0, 0, 0, 0, 0, 0 };
    __builtin_amdgcn_tensor_load_to_lds(g0, g1, z4, z4, z8, /*cpol=*/0);
}

// ---------------------------------------------------------------------------
// Prep 1: Wk2f = (Wk @ SK^T) * scale  (512x64), same for Wq2f; folded biases.
// ---------------------------------------------------------------------------
__global__ void smp_prep_proj(const float* __restrict__ Wk, const float* __restrict__ Wq,
                              const float* __restrict__ sk, const float* __restrict__ bk,
                              const float* __restrict__ bq, const float* __restrict__ scale,
                              float* __restrict__ Wk2f, float* __restrict__ Wq2f,
                              float* __restrict__ bk2, float* __restrict__ bq2) {
    int id = blockIdx.x * blockDim.x + threadIdx.x;
    float sc = scale[0];
    if (id < 2 * 512 * 64) {
        int sel = id >> 15;            // 0 -> Wk, 1 -> Wq
        int c   = (id & 32767) >> 6;   // input-dim row
        int s   = id & 63;             // slot
        const float* W = sel ? Wq : Wk;
        float acc = 0.f;
        for (int n = 0; n < 512; ++n) acc += W[c * 512 + n] * sk[s * 512 + n];
        (sel ? Wq2f : Wk2f)[c * 64 + s] = acc * sc;
    } else if (id < 2 * 512 * 64 + 128) {
        int id2 = id - 2 * 512 * 64;
        int sel = id2 >> 6, s = id2 & 63;
        const float* b = sel ? bq : bk;
        float acc = 0.f;
        for (int n = 0; n < 512; ++n) acc += b[n] * sk[s * 512 + n];
        (sel ? bq2 : bk2)[s] = acc * sc;
    }
}

// ---------------------------------------------------------------------------
// Prep 2: pack a row-major f32 (512 x N) weight into bf16 WMMA B-fragments:
// frag order [kstep][ntile], within frag: element = lane*16 + j,
// n = nt*16 + (lane&15), k = kstep*32 + (lane<16?0:8) + (j<8 ? j : j+8)
// ---------------------------------------------------------------------------
__global__ void smp_pack(const float* __restrict__ src, bf16_t* __restrict__ dst, int N) {
    int id = blockIdx.x * blockDim.x + threadIdx.x;
    int total = 512 * N;
    if (id >= total) return;
    int NT = N >> 4;
    int f = id >> 9;
    int e = id & 511;
    int kstep = f / NT, nt = f % NT;
    int lane = e >> 4, j = e & 15;
    int n = nt * 16 + (lane & 15);
    int k = kstep * 32 + ((lane < 16) ? 0 : 8) + ((j < 8) ? j : j + 8);
    dst[id] = (bf16_t)src[k * N + n];
}

// ---------------------------------------------------------------------------
// Main fused kernel: one workgroup per 64-token chunk.
// ---------------------------------------------------------------------------
__global__ __launch_bounds__(256, 1)
void smp_main(const float* __restrict__ x, const float* __restrict__ bv,
              const float* __restrict__ ln_g, const float* __restrict__ ln_b,
              const float* __restrict__ bo,
              const bf16_t* __restrict__ pWv, const bf16_t* __restrict__ pWo,
              const bf16_t* __restrict__ pWk2, const bf16_t* __restrict__ pWq2,
              const float* __restrict__ bk2, const float* __restrict__ bq2,
              float* __restrict__ out) {
    __shared__ alignas(16) char smem[SMEM_BYTES];
    bf16_t* xbf   = (bf16_t*)(smem + XBF_OFF);
    bf16_t* ww_bf = (bf16_t*)(smem + WWBF_OFF);
    bf16_t* rw_bf = (bf16_t*)(smem + RWBF_OFF);
    bf16_t* Rbf   = (bf16_t*)(smem + RBF_OFF);
    bf16_t* Vt    = (bf16_t*)(smem + VT_OFF);
    float*  lgww  = (float*)(smem + LGWW_OFF);
    float*  lgrw  = (float*)(smem + LGRW_OFF);
    bf16_t* tdm   = (bf16_t*)(smem + TDM_OFF);
    bf16_t* Abf   = (bf16_t*)(smem + ABF_OFF);
    float*  scr   = (float*)(smem + SCR_OFF);

    const int tid  = threadIdx.x;
    const int lane = tid & 31;
    const int wave = tid >> 5;
    const int mT   = wave >> 1;    // row tile (16 tokens)
    const int nh   = wave & 1;     // column half
    const int tok0 = blockIdx.x * TOK;
    const int rhalf = (lane < 16) ? 0 : 8;

    v8f zero = { 0.f, 0.f, 0.f, 0.f, 0.f, 0.f, 0.f, 0.f };
    v8f accV[16];
    v8f accK[2], accQ[2];
#pragma unroll
    for (int j = 0; j < 16; ++j) accV[j] = zero;
    accK[0] = accK[1] = accQ[0] = accQ[1] = zero;

    // ---- Stage 1: X in two K-halves; fused slot-logit + V projections ------
    for (int h = 0; h < 2; ++h) {
        // load & convert the 64x256 f32 half-chunk of x into LDS bf16
        for (int i = 0; i < 16; ++i) {
            int flat = tid + i * 256;            // 0..4095
            int t  = flat >> 6;
            int c4 = flat & 63;                  // group of 4 columns
            const v4f* xp = (const v4f*)(x + (size_t)(tok0 + t) * DIMD + h * 256 + c4 * 4);
            v4f v = *xp;
            v4bf b = { (bf16_t)v[0], (bf16_t)v[1], (bf16_t)v[2], (bf16_t)v[3] };
            *(v4bf*)(xbf + t * XBF_STRIDE + c4 * 4) = b;
        }
        __syncthreads();

        for (int kb = 0; kb < 8; ++kb) {
            int kstep = h * 8 + kb;
            v16bf a = load_afrag(xbf, XBF_STRIDE, mT * 16, kb * 32, lane);
#pragma unroll
            for (int j = 0; j < 2; ++j) {
                int nt = nh * 2 + j;
                v16bf bk_ = load_bfrag_packed(pWk2, kstep * 4 + nt, lane);
                accK[j] = wmma_bf16(a, bk_, accK[j]);
                v16bf bq_ = load_bfrag_packed(pWq2, kstep * 4 + nt, lane);
                accQ[j] = wmma_bf16(a, bq_, accQ[j]);
            }
#pragma unroll
            for (int j = 0; j < 16; ++j) {
                int nt = nh * 16 + j;
                v16bf bw = load_bfrag_packed(pWv, kstep * 32 + nt, lane);
                accV[j] = wmma_bf16(a, bw, accV[j]);
            }
        }
        __syncthreads();   // xbf reused for the next half
    }

    // ---- Stage 2: logits -> LDS, softmax over slots -------------------------
#pragma unroll
    for (int j = 0; j < 2; ++j) {
#pragma unroll
        for (int r = 0; r < 8; ++r) {
            int m = mT * 16 + r + rhalf;
            int n = (nh * 2 + j) * 16 + (lane & 15);
            lgww[m * 64 + n] = accK[j][r];
            lgrw[m * 64 + n] = accQ[j][r];
        }
    }
    __syncthreads();

    if (tid < 128) {
        int t = tid & 63, sel = tid >> 6;
        const float* src  = sel ? lgrw : lgww;
        const float* bias = sel ? bq2  : bk2;
        bf16_t* dst = sel ? rw_bf : ww_bf;
        float vmax = -1e30f;
        for (int s = 0; s < NSLOT; ++s) {
            float v = src[t * 64 + s] + bias[s];
            vmax = fmaxf(vmax, v);
        }
        float ssum = 0.f;
        for (int s = 0; s < NSLOT; ++s)
            ssum += __expf(src[t * 64 + s] + bias[s] - vmax);
        float inv = 1.f / ssum;
        for (int s = 0; s < NSLOT; ++s)
            dst[t * 72 + s] = (bf16_t)(__expf(src[t * 64 + s] + bias[s] - vmax) * inv);
    }
    __syncthreads();   // logits region now dead -> Vt may be written

    // ---- Stage 3: V + bias -> transposed LDS (Vt[d][t]) --------------------
#pragma unroll
    for (int j = 0; j < 16; ++j) {
        int d = (nh * 16 + j) * 16 + (lane & 15);
        float bvv = bv[d];
#pragma unroll
        for (int r = 0; r < 8; ++r) {
            int t = mT * 16 + r + rhalf;
            Vt[d * VT_STRIDE + t] = (bf16_t)(accV[j][r] + bvv);
        }
    }

    // ---- Stage 4: A = tril(rw @ ww^T) --------------------------------------
    v8f accA[2];
    accA[0] = accA[1] = zero;
    for (int ks = 0; ks < 2; ++ks) {
        v16bf a = load_afrag(rw_bf, 72, mT * 16, ks * 32, lane);
#pragma unroll
        for (int j = 0; j < 2; ++j) {
            int nt = nh * 2 + j;
            v16bf b = load_afrag(ww_bf, 72, nt * 16, ks * 32, lane);
            accA[j] = wmma_bf16(a, b, accA[j]);
        }
    }
#pragma unroll
    for (int j = 0; j < 2; ++j) {
#pragma unroll
        for (int r = 0; r < 8; ++r) {
            int m = mT * 16 + r + rhalf;
            int n = (nh * 2 + j) * 16 + (lane & 15);
            float v = (n <= m) ? accA[j][r] : 0.f;
            Abf[m * 72 + n] = (bf16_t)v;
        }
    }
    __syncthreads();   // Vt + Abf complete, ww/rw/x regions now dead

    // ---- Stage 5: R = A @ V -------------------------------------------------
    v8f accR[16];
#pragma unroll
    for (int j = 0; j < 16; ++j) accR[j] = zero;
    for (int ks = 0; ks < 2; ++ks) {
        v16bf a = load_afrag(Abf, 72, mT * 16, ks * 32, lane);
#pragma unroll
        for (int j = 0; j < 16; ++j) {
            int nt = nh * 16 + j;
            v16bf b = load_afrag(Vt, VT_STRIDE, nt * 16, ks * 32, lane);
            accR[j] = wmma_bf16(a, b, accR[j]);
        }
    }
#pragma unroll
    for (int j = 0; j < 16; ++j) {
#pragma unroll
        for (int r = 0; r < 8; ++r) {
            int t = mT * 16 + r + rhalf;
            int d = (nh * 16 + j) * 16 + (lane & 15);
            Rbf[t * RBF_STRIDE + d] = (bf16_t)accR[j][r];
        }
    }
    __syncthreads();   // R complete; Vt dead -> region becomes TDM staging

    // ---- Stage 6: LayerNorm in place ---------------------------------------
    {
        int row = tid >> 2, part = tid & 3;
        float s1 = 0.f, s2 = 0.f;
        for (int c = 0; c < 128; ++c) {
            float v = (float)Rbf[row * RBF_STRIDE + part * 128 + c];
            s1 += v; s2 += v * v;
        }
        scr[(row * 4 + part) * 2 + 0] = s1;
        scr[(row * 4 + part) * 2 + 1] = s2;
        __syncthreads();
        float mu = 0.f, m2 = 0.f;
        for (int p = 0; p < 4; ++p) {
            mu += scr[(row * 4 + p) * 2 + 0];
            m2 += scr[(row * 4 + p) * 2 + 1];
        }
        mu *= (1.f / 512.f);
        m2 = m2 * (1.f / 512.f) - mu * mu;
        float inv = rsqrtf(m2 + 1e-5f);
        for (int c = 0; c < 128; ++c) {
            int d = part * 128 + c;
            float v = (float)Rbf[row * RBF_STRIDE + d];
            Rbf[row * RBF_STRIDE + d] = (bf16_t)((v - mu) * inv * ln_g[d] + ln_b[d]);
        }
    }
    __syncthreads();

    // ---- Stage 7: out = LN @ Wo + bo + x, Wo streamed via TDM ---------------
    unsigned int lds_off = (unsigned int)(uintptr_t)tdm;  // low 32 bits = LDS byte offset
    if (wave == 0) tdm_load(lds_off, pWo, 8192);          // 32KB fragment block, kstep 0
    v8f accO[16];
#pragma unroll
    for (int j = 0; j < 16; ++j) accO[j] = zero;

    for (int ks = 0; ks < 16; ++ks) {
        if (wave == 0) __builtin_amdgcn_s_wait_tensorcnt(0);
        __syncthreads();
        v16bf a = load_afrag(Rbf, RBF_STRIDE, mT * 16, ks * 32, lane);
#pragma unroll
        for (int j = 0; j < 16; ++j) {
            int nt = nh * 16 + j;
            const v16bf* bp = (const v16bf*)(tdm + nt * 512);
            v16bf b = bp[lane];
            accO[j] = wmma_bf16(a, b, accO[j]);
        }
        __syncthreads();
        if (wave == 0 && ks < 15)
            tdm_load(lds_off, pWo + (size_t)(ks + 1) * 16384, 8192);
    }

#pragma unroll
    for (int j = 0; j < 16; ++j) {
        int d = (nh * 16 + j) * 16 + (lane & 15);
        float bov = bo[d];
#pragma unroll
        for (int r = 0; r < 8; ++r) {
            int t = mT * 16 + r + rhalf;
            size_t gi = (size_t)(tok0 + t) * DIMD + d;
            out[gi] = accO[j][r] + bov + x[gi];
        }
    }
}

// ---------------------------------------------------------------------------
extern "C" void kernel_launch(void* const* d_in, const int* in_sizes, int n_in,
                              void* d_out, int out_size, void* d_ws, size_t ws_size,
                              hipStream_t stream) {
    const float* x    = (const float*)d_in[0];
    const float* sk   = (const float*)d_in[1];
    const float* Wk   = (const float*)d_in[2];
    const float* bk   = (const float*)d_in[3];
    const float* Wq   = (const float*)d_in[4];
    const float* bq   = (const float*)d_in[5];
    const float* Wv   = (const float*)d_in[6];
    const float* bv   = (const float*)d_in[7];
    const float* sc   = (const float*)d_in[8];
    const float* ln_g = (const float*)d_in[9];
    const float* ln_b = (const float*)d_in[10];
    const float* Wo   = (const float*)d_in[11];
    const float* bo   = (const float*)d_in[12];
    float* out = (float*)d_out;

    char* ws = (char*)d_ws;                 // needs ~1.45 MB
    bf16_t* pWv  = (bf16_t*)(ws + WS_PWV);
    bf16_t* pWo  = (bf16_t*)(ws + WS_PWO);
    bf16_t* pWk2 = (bf16_t*)(ws + WS_PWK2);
    bf16_t* pWq2 = (bf16_t*)(ws + WS_PWQ2);
    float*  bk2  = (float*)(ws + WS_BK2);
    float*  bq2  = (float*)(ws + WS_BQ2);
    float*  Wk2f = (float*)(ws + WS_WK2F);
    float*  Wq2f = (float*)(ws + WS_WQ2F);

    smp_prep_proj<<<257, 256, 0, stream>>>(Wk, Wq, sk, bk, bq, sc, Wk2f, Wq2f, bk2, bq2);
    smp_pack<<<1024, 256, 0, stream>>>(Wv,   pWv,  512);
    smp_pack<<<1024, 256, 0, stream>>>(Wo,   pWo,  512);
    smp_pack<<<128,  256, 0, stream>>>(Wk2f, pWk2, 64);
    smp_pack<<<128,  256, 0, stream>>>(Wq2f, pWq2, 64);
    smp_main<<<NCHUNKS, 256, 0, stream>>>(x, bv, ln_g, ln_b, bo,
                                          pWv, pWo, pWk2, pWq2, bk2, bq2, out);
}
